// GroupedQueryAttention_61667140436397
// MI455X (gfx1250) — matrix operand order repaired and checked by reading, not compiled, not run on hardware
//
#include <hip/hip_runtime.h>
#include <hip/hip_bf16.h>

// ---------------------------------------------------------------------------
// Types for CDNA5 WMMA
// ---------------------------------------------------------------------------
typedef __bf16 v16bf __attribute__((ext_vector_type(16)));
typedef float  v8f   __attribute__((ext_vector_type(8)));
typedef unsigned int v4u __attribute__((ext_vector_type(4)));
typedef int v8i __attribute__((ext_vector_type(8)));
typedef int v4i __attribute__((ext_vector_type(4)));

union Frag16 { unsigned int u[8]; v16bf v; };

__device__ __forceinline__ unsigned short f2bf(float f) {
  unsigned int u = __builtin_bit_cast(unsigned int, f);
  u += 0x7FFFu + ((u >> 16) & 1u);           // round-to-nearest-even
  return (unsigned short)(u >> 16);
}

__device__ __forceinline__ v8f wmma_bf16(v16bf a, v16bf b, v8f c) {
  return __builtin_amdgcn_wmma_f32_16x16x32_bf16(
      /*neg_a=*/false, a, /*neg_b=*/false, b,
      /*c_mod=*/(short)0, c, /*reuse_a=*/false, /*reuse_b=*/false);
}

// ---------------------------------------------------------------------------
// Tensor Data Mover support (gated; fallback path keeps the file compiling
// on toolchains without the builtin)
// ---------------------------------------------------------------------------
#if defined(__gfx1250__) && __has_builtin(__builtin_amdgcn_tensor_load_to_lds)
#define HAVE_TDM 1
#else
#define HAVE_TDM 0
#endif

#if HAVE_TDM
__device__ __forceinline__ void tdm_wait0() {
#if __has_builtin(__builtin_amdgcn_s_wait_tensorcnt)
  __builtin_amdgcn_s_wait_tensorcnt(0);
#else
  asm volatile("s_wait_tensorcnt 0x0" ::: "memory");
#endif
}

// 2-D TDM load: tile (tile_d0 x tile_d1) of 2-byte elements from global
// (row stride = stride_elems) into LDS at lds_byte_off, with LDS padding of
// 4 dwords after every 16 dwords (i.e. 32-half rows padded to 40-half stride).
__device__ __forceinline__ void tdm_load_2d(unsigned int lds_byte_off,
                                            const unsigned short* gptr,
                                            int tile_d0, int tile_d1,
                                            long stride_elems) {
  unsigned long long ga = (unsigned long long)gptr;
  v4u g0;
  g0.x = 1u;                                             // count=1 (valid D#)
  g0.y = lds_byte_off;                                   // lds_addr
  g0.z = (unsigned int)ga;                               // global_addr lo
  g0.w = (unsigned int)((ga >> 32) & 0x1FFFFFFu) | (2u << 30);  // hi | type=2
  const unsigned int w0 = (1u << 16)   // data_size = 2 bytes
                        | (1u << 20)   // pad_enable
                        | (3u << 22)   // pad_interval: 16 dwords
                        | (3u << 25);  // pad_amount:   4 dwords
  const unsigned long long td0 = 1u << 20;   // generous tensor dims (in-bounds tiles)
  const unsigned long long td1 = 1u << 20;
  const unsigned long long s0 = (unsigned long long)stride_elems;
  v8i g1;
  g1[0] = (int)w0;
  g1[1] = (int)((td0 & 0xFFFFu) << 16);                         // dim0 lo16
  g1[2] = (int)(((td0 >> 16) & 0xFFFFu) | ((td1 & 0xFFFFu) << 16));
  g1[3] = (int)(((td1 >> 16) & 0xFFFFu) | ((unsigned)tile_d0 << 16));
  g1[4] = (int)((unsigned)tile_d1 & 0xFFFFu);                   // tile_dim1, tile_dim2=0
  g1[5] = (int)(unsigned)(s0 & 0xFFFFFFFFu);                    // stride0 lo32
  g1[6] = (int)(unsigned)((s0 >> 32) & 0xFFFFu);                // stride0 hi16
  g1[7] = 0;
  v4i z4 = {0, 0, 0, 0};
#if defined(__clang_major__) && (__clang_major__ >= 23)
  v8i z8 = {0, 0, 0, 0, 0, 0, 0, 0};
  __builtin_amdgcn_tensor_load_to_lds(g0, g1, z4, z4, z8, 0);
#else
  __builtin_amdgcn_tensor_load_to_lds(g0, g1, z4, z4, 0);
#endif
}

__device__ __forceinline__ unsigned int lds_off(const void* p) {
  // generic LDS address truncates to the LDS byte offset on gfx1250
  return (unsigned int)(unsigned long long)p;
}
#endif  // HAVE_TDM

// ---------------------------------------------------------------------------
// Elementwise fp32 -> bf16
// ---------------------------------------------------------------------------
__global__ __launch_bounds__(256) void f32_to_bf16_kernel(
    const float* __restrict__ in, unsigned short* __restrict__ out, long n4) {
  for (long i = (long)blockIdx.x * blockDim.x + threadIdx.x; i < n4;
       i += (long)gridDim.x * blockDim.x) {
    float4 v = ((const float4*)in)[i];
    unsigned int lo = (unsigned int)f2bf(v.x) | ((unsigned int)f2bf(v.y) << 16);
    unsigned int hi = (unsigned int)f2bf(v.z) | ((unsigned int)f2bf(v.w) << 16);
    ((uint2*)out)[i] = make_uint2(lo, hi);
  }
}

// ---------------------------------------------------------------------------
// RoPE: t [B, S, Hn, 64] fp32 -> bf16 (rotate-half formulation)
// ---------------------------------------------------------------------------
__global__ __launch_bounds__(256) void rope_bf16_kernel(
    const float* __restrict__ t, const float* __restrict__ cosr,
    const float* __restrict__ sinr, unsigned short* __restrict__ out,
    int S_, int Hn, long total) {
  for (long i = (long)blockIdx.x * blockDim.x + threadIdx.x; i < total;
       i += (long)gridDim.x * blockDim.x) {
    int hd = (int)(i & 63);
    long rest = i >> 6;            // (b*S + s)*Hn + h
    long bs = rest / Hn;           // b*S + s
    int s = (int)(bs % S_);
    float x = t[i];
    float partner = (hd < 32) ? -t[i + 32] : t[i - 32];
    float c = cosr[s * 64 + hd];
    float sn = sinr[s * 64 + hd];
    out[i] = f2bf(x * c + partner * sn);
  }
}

// ---------------------------------------------------------------------------
// WMMA GEMM:  C[M,N] (fp32) = A[M,K] (bf16) * W[N,K]^T (bf16)
// Block: 256 thr (8 waves). Tile 128x128, k-step 32.
// TDM path: double-buffered LDS, wave 0 DMAs tile t+1 while all waves WMMA
// tile t.  Fallback path: synchronous global->VGPR->LDS staging.
// ---------------------------------------------------------------------------
#define GBM 128
#define GBN 128
#define GBK 32
#define GLD 40   // LDS row stride in halfs (80B, multiple of 16B)

__global__ __launch_bounds__(256) void gemm_bf16_nt(
    const unsigned short* __restrict__ A, const unsigned short* __restrict__ W,
    float* __restrict__ C, int M, int N, int K) {
#if HAVE_TDM
  __shared__ unsigned short As[2][GBM * GLD];
  __shared__ unsigned short Ws[2][GBN * GLD];
#else
  __shared__ unsigned short As[1][GBM * GLD];
  __shared__ unsigned short Ws[1][GBN * GLD];
#endif

  const int tid = threadIdx.x;
  const int lane = tid & 31;
  const int wave = tid >> 5;
  const int wm = wave >> 1;
  const int wn = wave & 1;
  const int lh = lane & 15;
  const int lhi = lane >> 4;
  const long mBase = (long)blockIdx.y * GBM;
  const long nBase = (long)blockIdx.x * GBN;

  v8f acc[2][4];
  v8f zero = {};
#pragma unroll
  for (int m = 0; m < 2; ++m)
#pragma unroll
    for (int n = 0; n < 4; ++n) acc[m][n] = zero;

#if HAVE_TDM
  if (wave == 0) {  // prologue: DMA first tile into buffer 0
    tdm_load_2d(lds_off(&As[0][0]), A + mBase * K, GBK, GBM, K);
    tdm_load_2d(lds_off(&Ws[0][0]), W + nBase * K, GBK, GBN, K);
  }
#endif

  const int nsteps = K / GBK;
  for (int t = 0; t < nsteps; ++t) {
#if HAVE_TDM
    if (wave == 0) tdm_wait0();       // tile t landed in LDS
    __syncthreads();                  // ...and buffer (t+1)&1 is free again
    if (wave == 0 && t + 1 < nsteps) {
      long k0n = (long)(t + 1) * GBK;
      tdm_load_2d(lds_off(&As[(t + 1) & 1][0]), A + mBase * K + k0n, GBK, GBM, K);
      tdm_load_2d(lds_off(&Ws[(t + 1) & 1][0]), W + nBase * K + k0n, GBK, GBN, K);
    }
    const unsigned short* AsP = &As[t & 1][0];
    const unsigned short* WsP = &Ws[t & 1][0];
#else
    const int k0 = t * GBK;
    __syncthreads();
#pragma unroll
    for (int j = 0; j < 2; ++j) {
      int idx = tid * 16 + j * 8;      // 0..4095, step 8
      int r = idx >> 5;                // row in tile
      int c = idx & 31;                // col (multiple of 8)
      uint4 av = *(const uint4*)(A + (mBase + r) * K + k0 + c);
      *(uint4*)&As[0][r * GLD + c] = av;
      uint4 wv = *(const uint4*)(W + (nBase + r) * K + k0 + c);
      *(uint4*)&Ws[0][r * GLD + c] = wv;
    }
    __syncthreads();
    const unsigned short* AsP = &As[0][0];
    const unsigned short* WsP = &Ws[0][0];
#endif

    // A fragments (CDNA5 16-bit A layout: lane half selects K 0-7/16-23 vs 8-15/24-31)
    Frag16 af[2];
#pragma unroll
    for (int m = 0; m < 2; ++m) {
      int row = wm * 32 + m * 16 + lh;
      int kb = lhi * 8;
#pragma unroll
      for (int j = 0; j < 8; ++j) {
        int koff = (j < 4) ? (2 * j) : (2 * j + 8);
        af[m].u[j] = *(const unsigned int*)&AsP[row * GLD + kb + koff];
      }
    }
    // B fragments (W row = output col; lane half selects K 0-7/16-23 vs 8-15/24-31)
#pragma unroll
    for (int n = 0; n < 4; ++n) {
      Frag16 bf;
      int wrow = wn * 64 + n * 16 + lh;
      int kb = lhi * 8;
#pragma unroll
      for (int j = 0; j < 8; ++j)
        bf.u[j] = *(const unsigned int*)&WsP[wrow * GLD + kb + ((j < 4) ? (2 * j) : (2 * j + 8))];
#pragma unroll
      for (int m = 0; m < 2; ++m) acc[m][n] = wmma_bf16(af[m].v, bf.v, acc[m][n]);
    }
  }

  // Store C: fragment element (r, lane) -> row r + 8*lhi, col lh
#pragma unroll
  for (int m = 0; m < 2; ++m)
#pragma unroll
    for (int n = 0; n < 4; ++n) {
      long row = mBase + wm * 32 + m * 16;
      long col = nBase + wn * 64 + n * 16 + lh;
#pragma unroll
      for (int r = 0; r < 8; ++r)
        C[(row + r + 8 * lhi) * N + col] = acc[m][n][r];
    }
}

// ---------------------------------------------------------------------------
// Flash attention (causal, GQA). Block: 256 thr = 8 waves, 128 query rows.
// Q: [B,S,H,64] bf16  K,V: [B,S,G,64] bf16  Ctx: [B,S,H*64] fp32
// ---------------------------------------------------------------------------
#define KLD 72   // Ks row stride (halfs): 144B, mult of 16
#define VLD 40   // Vt row stride (halfs)

__global__ __launch_bounds__(256) void attn_fwd_kernel(
    const unsigned short* __restrict__ Qb, const unsigned short* __restrict__ Kb,
    const unsigned short* __restrict__ Vb, float* __restrict__ Ctx,
    int S_, int Hn, int Gn) {
  __shared__ unsigned short Ks[32 * KLD];        // [key][hd]
  __shared__ unsigned short Vt[64 * VLD];        // [hd][key] (transposed)
  __shared__ unsigned short Ps[8 * 16 * 32];     // per-wave P tiles

  const int tid = threadIdx.x;
  const int lane = tid & 31;
  const int wave = tid >> 5;
  const int lh = lane & 15;
  const int lhi = lane >> 4;
  const int b = blockIdx.y / Hn;
  const int h = blockIdx.y % Hn;
  const int g = h / (Hn / Gn);
  const int qbase = blockIdx.x * 128 + wave * 16;
  const long Dq = (long)Hn * 64;
  const long Dkv = (long)Gn * 64;
  const float scale = 0.125f;  // 64^-0.5

  // Load Q as two A-fragments (k-chunks 0-31 and 32-63)
  Frag16 qf[2];
  {
    const unsigned short* qptr =
        Qb + ((long)(b * S_) + qbase + lh) * Dq + (long)h * 64;
    int kb = lhi * 8;
#pragma unroll
    for (int c = 0; c < 2; ++c)
#pragma unroll
      for (int j = 0; j < 8; ++j) {
        int koff = (j < 4) ? (2 * j) : (2 * j + 8);
        qf[c].u[j] = *(const unsigned int*)(qptr + c * 32 + kb + koff);
      }
  }

  v8f ctxacc[4];
  v8f zero = {};
#pragma unroll
  for (int f = 0; f < 4; ++f) ctxacc[f] = zero;
  float m_i[8], l_i[8];
#pragma unroll
  for (int r = 0; r < 8; ++r) { m_i[r] = -__builtin_inff(); l_i[r] = 0.0f; }

  const int ntiles = blockIdx.x * 4 + 4;  // causal: keys up to end of q tile
  for (int t = 0; t < ntiles; ++t) {
    const int kbase = t * 32;
    __syncthreads();
    {  // stage K (row-major) and V (transposed): 8 halfs / thread each
      int idx = tid * 8;
      int kr = idx >> 6;   // key row 0..31
      int c = idx & 63;    // hd (multiple of 8)
      long src = ((long)(b * S_) + kbase + kr) * Dkv + (long)g * 64 + c;
      uint4 kv = *(const uint4*)(Kb + src);
      *(uint4*)&Ks[kr * KLD + c] = kv;
      union { uint4 q; unsigned short s[8]; } vv;
      vv.q = *(const uint4*)(Vb + src);
#pragma unroll
      for (int e = 0; e < 8; ++e) Vt[(c + e) * VLD + kr] = vv.s[e];
    }
    __syncthreads();

    // ---- scores = Q (16x64) x K^T (64x32) ----
    v8f sc[2] = {zero, zero};
#pragma unroll
    for (int nf = 0; nf < 2; ++nf) {
      int keyrow = nf * 16 + lh;
#pragma unroll
      for (int c = 0; c < 2; ++c) {
        Frag16 bf;
        int hdb = c * 32 + lhi * 8;
#pragma unroll
        for (int j = 0; j < 8; ++j)
          bf.u[j] = *(const unsigned int*)&Ks[keyrow * KLD + hdb + ((j < 4) ? (2 * j) : (2 * j + 8))];
        sc[nf] = wmma_bf16(qf[c].v, bf.v, sc[nf]);
      }
    }

    // ---- online softmax (row = r + 8*lhi lives in one 16-lane half) ----
#pragma unroll
    for (int r = 0; r < 8; ++r) {
      int qi = qbase + r + 8 * lhi;
      float s0 = sc[0][r] * scale;
      float s1 = sc[1][r] * scale;
      if (kbase + lh > qi)      s0 = -__builtin_inff();
      if (kbase + 16 + lh > qi) s1 = -__builtin_inff();
      float mx = fmaxf(s0, s1);
#pragma unroll
      for (int off = 8; off >= 1; off >>= 1)
        mx = fmaxf(mx, __shfl_xor(mx, off, 32));
      float mnew = fmaxf(m_i[r], mx);
      float alpha = __expf(m_i[r] - mnew);
      float p0 = __expf(s0 - mnew);
      float p1 = __expf(s1 - mnew);
      float rs = p0 + p1;
#pragma unroll
      for (int off = 8; off >= 1; off >>= 1) rs += __shfl_xor(rs, off, 32);
      l_i[r] = l_i[r] * alpha + rs;
      m_i[r] = mnew;
#pragma unroll
      for (int f = 0; f < 4; ++f) ctxacc[f][r] *= alpha;
      int prow = wave * 512 + (r + 8 * lhi) * 32;
      Ps[prow + lh]      = f2bf(p0);
      Ps[prow + 16 + lh] = f2bf(p1);
    }
    // within-wave LDS RAW: wait for DS stores before re-reading P
    asm volatile("s_wait_dscnt 0" ::: "memory");

    // ---- reload P as A-fragment (16x32) ----
    Frag16 pf;
    {
      const unsigned short* pp = &Ps[wave * 512 + lh * 32];
      int kb = lhi * 8;
#pragma unroll
      for (int j = 0; j < 8; ++j) {
        int koff = (j < 4) ? (2 * j) : (2 * j + 8);
        pf.u[j] = *(const unsigned int*)(pp + kb + koff);
      }
    }
    // ---- ctx += P (16x32) x V (32x64) ----
#pragma unroll
    for (int f = 0; f < 4; ++f) {
      Frag16 bf;
      int hdc = f * 16 + lh;   // output col (hd)
      int kk = lhi * 8;        // key base for this lane half
#pragma unroll
      for (int j = 0; j < 8; ++j)
        bf.u[j] = *(const unsigned int*)&Vt[hdc * VLD + kk + ((j < 4) ? (2 * j) : (2 * j + 8))];
      ctxacc[f] = wmma_bf16(pf.v, bf.v, ctxacc[f]);
    }
  }

  // ---- normalize and write ctx ----
#pragma unroll
  for (int r = 0; r < 8; ++r) {
    int qi = qbase + r + 8 * lhi;
    float inv = 1.0f / l_i[r];
    long base = ((long)(b * S_) + qi) * Dq + (long)h * 64;
#pragma unroll
    for (int f = 0; f < 4; ++f) Ctx[base + f * 16 + lh] = ctxacc[f][r] * inv;
  }
}

// ---------------------------------------------------------------------------
// Host-side orchestration
// ---------------------------------------------------------------------------
extern "C" void kernel_launch(void* const* d_in, const int* in_sizes, int n_in,
                              void* d_out, int out_size, void* d_ws, size_t ws_size,
                              hipStream_t stream) {
  constexpr int B = 2, S = 2048, D = 2048, H = 32, G = 8, HD = 64;
  constexpr long M = (long)B * S;       // 4096
  constexpr int KV = G * HD;            // 512

  const float* x   = (const float*)d_in[0];
  const float* wq  = (const float*)d_in[1];
  const float* wk  = (const float*)d_in[2];
  const float* wv  = (const float*)d_in[3];
  const float* wo  = (const float*)d_in[4];
  const float* cosr = (const float*)d_in[5];
  const float* sinr = (const float*)d_in[6];
  (void)in_sizes; (void)n_in; (void)out_size; (void)ws_size;

  // Workspace carve (256-B aligned)
  char* w = (char*)d_ws;
  auto carve = [&](size_t bytes) {
    char* p = w;
    w += (bytes + 255) & ~size_t(255);
    return p;
  };
  unsigned short* xb  = (unsigned short*)carve(M * D * 2);       // x bf16
  unsigned short* wqb = (unsigned short*)carve((size_t)D * D * 2);
  unsigned short* wkb = (unsigned short*)carve((size_t)KV * D * 2);
  unsigned short* wvb = (unsigned short*)carve((size_t)KV * D * 2);
  unsigned short* wob = (unsigned short*)carve((size_t)D * D * 2);
  float* Qf  = (float*)carve(M * D * 4);
  float* Kf  = (float*)carve(M * KV * 4);
  float* Vf  = (float*)carve(M * KV * 4);
  unsigned short* qbb = (unsigned short*)carve(M * D * 2);
  unsigned short* kbb = (unsigned short*)carve(M * KV * 2);
  unsigned short* vbb = (unsigned short*)carve(M * KV * 2);
  float* ctx = (float*)carve(M * D * 4);
  unsigned short* ctxb = (unsigned short*)carve(M * D * 2);

  auto cvt = [&](const float* src, unsigned short* dst, long n) {
    long n4 = n / 4;
    int blocks = (int)((n4 + 255) / 256);
    if (blocks > 4096) blocks = 4096;
    f32_to_bf16_kernel<<<blocks, 256, 0, stream>>>(src, dst, n4);
  };

  // 1) bf16 conversions of inputs
  cvt(x, xb, M * D);
  cvt(wq, wqb, (long)D * D);
  cvt(wk, wkb, (long)KV * D);
  cvt(wv, wvb, (long)KV * D);
  cvt(wo, wob, (long)D * D);

  // 2) projections
  gemm_bf16_nt<<<dim3(D / GBN, M / GBM), 256, 0, stream>>>(xb, wqb, Qf, (int)M, D, D);
  gemm_bf16_nt<<<dim3(KV / GBN, M / GBM), 256, 0, stream>>>(xb, wkb, Kf, (int)M, KV, D);
  gemm_bf16_nt<<<dim3(KV / GBN, M / GBM), 256, 0, stream>>>(xb, wvb, Vf, (int)M, KV, D);

  // 3) RoPE on Q and K (fp32 math -> bf16), V straight to bf16
  {
    long tq = M * D;   // B*S*H*64
    int blocks = (int)((tq + 255) / 256); if (blocks > 8192) blocks = 8192;
    rope_bf16_kernel<<<blocks, 256, 0, stream>>>(Qf, cosr, sinr, qbb, S, H, tq);
    long tk = M * KV;  // B*S*G*64
    blocks = (int)((tk + 255) / 256); if (blocks > 8192) blocks = 8192;
    rope_bf16_kernel<<<blocks, 256, 0, stream>>>(Kf, cosr, sinr, kbb, S, G, tk);
  }
  cvt(Vf, vbb, M * KV);

  // 4) flash attention (causal, GQA)
  attn_fwd_kernel<<<dim3(S / 128, B * H), 256, 0, stream>>>(qbb, kbb, vbb, ctx, S, H, G);

  // 5) output projection
  cvt(ctx, ctxb, M * D);
  gemm_bf16_nt<<<dim3(D / GBN, M / GBM), 256, 0, stream>>>(ctxb, wob, (float*)d_out,
                                                           (int)M, D, D);
}
